// SelfAttentionBlock_90761248899400
// MI455X (gfx1250) — matrix operand order repaired and checked by reading, not compiled, not run on hardware
//
#include <hip/hip_runtime.h>

// ---------------------------------------------------------------------------
// Sizes fixed by the reference: B=8, N=2048, DIM=512, 3*DIM=1536, scale 1/sqrt(64)
// ---------------------------------------------------------------------------
constexpr int B_    = 8;
constexpr int N_    = 2048;
constexpr int DIM_  = 512;
constexpr int TDIM_ = 3 * DIM_;   // 1536

typedef __attribute__((ext_vector_type(8)))  float   v8f;
typedef __attribute__((ext_vector_type(8)))  __bf16  v8bf;
typedef __attribute__((ext_vector_type(16))) __bf16  v16bf;

__device__ __forceinline__ v16bf cat16(v8bf lo, v8bf hi) {
  v16bf r;
#pragma unroll
  for (int i = 0; i < 8; ++i) { r[i] = lo[i]; r[i + 8] = hi[i]; }
  return r;
}

__device__ __forceinline__ v8f wmma_bf16(v16bf a, v16bf b, v8f c) {
  // (neg_a, A, neg_b, B, c_mod, C, reuse_a, reuse_b)
  return __builtin_amdgcn_wmma_f32_16x16x32_bf16(false, a, false, b,
                                                 (short)0, c, false, false);
}

// A operand (16x32 bf16) from a row-major [16 x rowStride] buffer, K-offset k0.
__device__ __forceinline__ v16bf load_A(const __bf16* base, int rowStride, int k0) {
  const int lane = threadIdx.x & 31;
  const int hi   = lane >> 4;
  const int row  = lane & 15;
  const __bf16* p = base + row * rowStride + k0 + 8 * hi;
  v8bf lo = *(const v8bf*)p;
  v8bf hh = *(const v8bf*)(p + 16);
  return cat16(lo, hh);
}

// B operand (32x16 bf16): column n of B is row n of `base` (row-major).
__device__ __forceinline__ v16bf load_B(const __bf16* base, long rowStride, int k0) {
  const int lane = threadIdx.x & 31;
  const int hi   = lane >> 4;
  const int col  = lane & 15;
  const __bf16* p = base + (long)col * rowStride + k0 + 8 * hi;
  v8bf lo = *(const v8bf*)p;
  v8bf hh = *(const v8bf*)(p + 16);
  return cat16(lo, hh);
}

// ---------------------------------------------------------------------------
// CDNA5 async global->LDS copy (ASYNCcnt-tracked, no staging VGPRs).
// LDS operand is the low 32 bits of the generic pointer (LDS aperture keeps the
// allocation byte-offset in addr[31:0]).
// ---------------------------------------------------------------------------
__device__ __forceinline__ void async_copy_b128(unsigned lds_byte_addr,
                                                const void* gaddr) {
  asm volatile("global_load_async_to_lds_b128 %0, %1, off"
               :: "v"(lds_byte_addr),
                  "v"((unsigned long long)(uintptr_t)gaddr)
               : "memory");
}
__device__ __forceinline__ void wait_async0() {
  asm volatile("s_wait_asynccnt 0" ::: "memory");
}

// Stage `bytes` (multiple of 16*nthreads) from global to LDS, all threads.
__device__ __forceinline__ void stage_to_lds(__bf16* lds, const __bf16* gsrc,
                                             int bytes, int nthreads) {
  const unsigned ldsBase = (unsigned)(uintptr_t)lds;
  const char* src = (const char*)gsrc;
  for (int i = (int)threadIdx.x; i < bytes / 16; i += nthreads)
    async_copy_b128(ldsBase + (unsigned)i * 16u, src + (size_t)i * 16u);
  wait_async0();
}

// ---------------------------------------------------------------------------
// Kernel 1: degree gating.  One 256-thread block per (b,n) row:
//   s = sum_m rel_pos[b,n,m];  xg[d] = bf16( x[d] * sigmoid(s*w_d[d]+b_d[d]) )
// ---------------------------------------------------------------------------
__global__ __launch_bounds__(256) void deg_gate_kernel(
    const float* __restrict__ x, const float* __restrict__ rel,
    const float* __restrict__ wd, const float* __restrict__ bd,
    __bf16* __restrict__ xg) {
  const int bn  = blockIdx.x;      // [0, B*N)
  const int tid = threadIdx.x;
  const float* r = rel + (size_t)bn * N_;
  float s = 0.f;
#pragma unroll
  for (int k = 0; k < N_ / 256; ++k) s += r[tid + 256 * k];
  __shared__ float red[256];
  red[tid] = s;
  __syncthreads();
  for (int off = 128; off > 0; off >>= 1) {
    if (tid < off) red[tid] += red[tid + off];
    __syncthreads();
  }
  const float deg_in = red[0];
  const float* xr = x + (size_t)bn * DIM_;
  __bf16* og = xg + (size_t)bn * DIM_;
#pragma unroll
  for (int k = 0; k < DIM_ / 256; ++k) {
    const int d = tid + 256 * k;
    const float t = deg_in * wd[d] + bd[d];
    const float g = 1.f / (1.f + __expf(-t));
    og[d] = (__bf16)(xr[d] * g);
  }
}

// ---------------------------------------------------------------------------
// Kernel 2: w_qkv [512,1536] f32 -> wT [1536,512] bf16, LDS-tiled so both the
// global read and the global write are coalesced (32x33 padded tile).
// ---------------------------------------------------------------------------
__global__ __launch_bounds__(256) void wq_transpose_kernel(
    const float* __restrict__ w, __bf16* __restrict__ wT) {
  const int tj = blockIdx.x % (TDIM_ / 32);   // 48 tile-cols (j)
  const int td = blockIdx.x / (TDIM_ / 32);   // 16 tile-rows (d)
  __shared__ float tile[32][33];
  const int tx = threadIdx.x & 31;
  const int ty = threadIdx.x >> 5;            // 8 rows per pass
#pragma unroll
  for (int r = 0; r < 32; r += 8)
    tile[ty + r][tx] = w[(size_t)(td * 32 + ty + r) * TDIM_ + tj * 32 + tx];
  __syncthreads();
#pragma unroll
  for (int r = 0; r < 32; r += 8)
    wT[(size_t)(tj * 32 + ty + r) * DIM_ + td * 32 + tx] =
        (__bf16)tile[tx][ty + r];
}

// ---------------------------------------------------------------------------
// Kernel 3: QKV projection. One block = 16 rows of one batch, 8 waves.
// Each wave owns 12 column tiles (192 of 1536 outputs). A staged in LDS async.
// Outputs: sigmoid(qk) bf16 row-major; V bf16 transposed [dim][N]; res f32.
// ---------------------------------------------------------------------------
__global__ __launch_bounds__(256) void qkv_kernel(
    const __bf16* __restrict__ xg, const __bf16* __restrict__ wT,
    const float* __restrict__ bqkv,
    __bf16* __restrict__ qks, __bf16* __restrict__ vt,
    float* __restrict__ res) {
  const int blk = blockIdx.x;             // [0, B*N/16)
  const int b   = blk / (N_ / 16);
  const int n0  = (blk % (N_ / 16)) * 16;

  __shared__ __bf16 Ash[16 * DIM_];       // 16 KiB
  stage_to_lds(Ash, xg + ((size_t)b * N_ + n0) * DIM_, 16 * DIM_ * 2, 256);
  __syncthreads();

  const int wave = threadIdx.x >> 5;
  const int lane = threadIdx.x & 31;
  const int hi = lane >> 4, col = lane & 15;

  for (int tt = 0; tt < TDIM_ / (16 * 8); ++tt) {        // 12 tiles / wave
    const int j0 = (wave * (TDIM_ / (16 * 8)) + tt) * 16;
    v8f acc = {};
    const __bf16* Bbase = wT + (size_t)j0 * DIM_;
#pragma unroll
    for (int k0 = 0; k0 < DIM_; k0 += 32)
      acc = wmma_bf16(load_A(Ash, DIM_, k0), load_B(Bbase, DIM_, k0), acc);

    const int j = j0 + col;
    const float bias = bqkv[j];
#pragma unroll
    for (int i = 0; i < 8; ++i) {
      const int n = n0 + i + 8 * hi;
      const float v = acc[i] + bias;
      if (j0 < DIM_) {                                   // qk -> sigmoid, bf16
        const float s = 1.f / (1.f + __expf(-v));
        qks[((size_t)b * N_ + n) * DIM_ + j] = (__bf16)s;
      } else if (j0 < 2 * DIM_) {                        // value -> bf16, transposed
        vt[((size_t)b * DIM_ + (j - DIM_)) * N_ + n] = (__bf16)v;
      } else {                                           // residual -> f32
        res[((size_t)b * N_ + n) * DIM_ + (j - 2 * DIM_)] = v;
      }
    }
  }
}

// ---------------------------------------------------------------------------
// Kernel 4: streaming attention.  One block = (b, 16-row tile), 4 waves.
// Per 128-m chunk: each wave builds its 16x32 score slab (16 WMMAs/16x16 tile
// over K=512), applies 0.125*rel_pos, accumulates rowsum partials in REGISTERS,
// bf16 -> LDS; barrier; all waves run O += S * V over K=128 on their private
// 128-column slice.  Rowsum is reduced once at the end (8 LDS atomics/lane
// total instead of 256); linear normalization in the epilogue:
//   out = relu(O/(rowsum+eps) + res) * scale.
// ---------------------------------------------------------------------------
__global__ __launch_bounds__(128) void attn_kernel(
    const __bf16* __restrict__ qks, const __bf16* __restrict__ vt,
    const float* __restrict__ res, const float* __restrict__ rel,
    const float* __restrict__ scale, float* __restrict__ out) {
  const int blk = blockIdx.x;             // [0, B*N/16)
  const int b   = blk / (N_ / 16);
  const int n0  = (blk % (N_ / 16)) * 16;

  __shared__ __bf16 Ash[16 * DIM_];       // qk row block, 16 KiB
  __shared__ __bf16 Ssh[16 * 128];        // score slab, 4 KiB
  __shared__ float  rowsum[16];

  stage_to_lds(Ash, qks + ((size_t)b * N_ + n0) * DIM_, 16 * DIM_ * 2, 128);
  if (threadIdx.x < 16) rowsum[threadIdx.x] = 0.f;
  __syncthreads();

  const int wave = threadIdx.x >> 5;
  const int lane = threadIdx.x & 31;
  const int hi = lane >> 4, col = lane & 15;

  v8f accO[8];
  float rsp[8];                           // per-lane rowsum partials
  {
    v8f z = {};
#pragma unroll
    for (int i = 0; i < 8; ++i) { accO[i] = z; rsp[i] = 0.f; }
  }

  const float* relBase = rel + ((size_t)b * N_ + n0) * N_;   // [16][N]

  for (int mc = 0; mc < N_; mc += 128) {
    if (mc + 128 < N_) {   // prefetch next rel_pos chunk (global_prefetch_b8)
      const float* pf = relBase + (size_t)(threadIdx.x & 15) * N_ +
                        mc + 128 + 8 * (threadIdx.x >> 4);
      __builtin_prefetch(pf, 0, 0);
    }

    // ---- score phase: this wave's 32 m-columns of the chunk
#pragma unroll
    for (int t = 0; t < 2; ++t) {
      const int m0 = mc + 32 * wave + 16 * t;
      const __bf16* Bbase = qks + ((size_t)b * N_ + m0) * DIM_;
      v8f accS = {};
#pragma unroll
      for (int k0 = 0; k0 < DIM_; k0 += 32)
        accS = wmma_bf16(load_A(Ash, DIM_, k0), load_B(Bbase, DIM_, k0), accS);

      const int sc = 32 * wave + 16 * t + col;
#pragma unroll
      for (int i = 0; i < 8; ++i) {
        const int row = i + 8 * hi;
        const float v = accS[i] * 0.125f * relBase[(size_t)row * N_ + m0 + col];
        rsp[i] += v;                       // register rowsum partial
        Ssh[row * 128 + sc] = (__bf16)v;
      }
    }
    __syncthreads();

    // ---- output phase: O[16,128-slice] += S[16,128] * V[128,128-slice]
#pragma unroll
    for (int k0 = 0; k0 < 128; k0 += 32) {
      const v16bf a = load_A(Ssh, 128, k0);
#pragma unroll
      for (int ct = 0; ct < 8; ++ct) {
        const int c0 = wave * 128 + ct * 16;
        const __bf16* Bbase = vt + ((size_t)b * DIM_ + c0) * N_ + mc;
        accO[ct] = wmma_bf16(a, load_B(Bbase, N_, k0), accO[ct]);
      }
    }
    __syncthreads();   // protect Ssh before next chunk overwrites
  }

  // ---- fold per-lane rowsum partials into LDS (once), then normalize
#pragma unroll
  for (int i = 0; i < 8; ++i) atomicAdd(&rowsum[i + 8 * hi], rsp[i]);
  __syncthreads();

  float rs[8];
#pragma unroll
  for (int i = 0; i < 8; ++i) rs[i] = rowsum[i + 8 * hi] + 1e-6f;
#pragma unroll
  for (int ct = 0; ct < 8; ++ct) {
    const int c = wave * 128 + ct * 16 + col;
    const float sc = scale[c];
#pragma unroll
    for (int i = 0; i < 8; ++i) {
      const int n = n0 + i + 8 * hi;
      float v = accO[ct][i] / rs[i] + res[((size_t)b * N_ + n) * DIM_ + c];
      v = v > 0.f ? v : 0.f;
      out[((size_t)b * N_ + n) * DIM_ + c] = v * sc;
    }
  }
}

// ---------------------------------------------------------------------------
// Host entry
// ---------------------------------------------------------------------------
extern "C" void kernel_launch(void* const* d_in, const int* in_sizes, int n_in,
                              void* d_out, int out_size, void* d_ws, size_t ws_size,
                              hipStream_t stream) {
  const float* x     = (const float*)d_in[0];
  const float* rel   = (const float*)d_in[1];
  const float* wqkv  = (const float*)d_in[2];
  const float* bqkv  = (const float*)d_in[3];
  const float* wd    = (const float*)d_in[4];
  const float* bd    = (const float*)d_in[5];
  const float* scale = (const float*)d_in[6];
  float* out = (float*)d_out;

  // Workspace layout (all fully written before read each launch):
  //   xg  bf16 [B,N,DIM]      16 MiB @ 0
  //   wT  bf16 [3DIM,DIM]    1.5 MiB @ 16 MiB
  //   qks bf16 [B,N,DIM]      16 MiB @ 18 MiB
  //   vt  bf16 [B,DIM,N]      16 MiB @ 34 MiB
  //   res f32  [B,N,DIM]      32 MiB @ 50 MiB   (total 82 MiB)
  char* ws = (char*)d_ws;
  __bf16* xg  = (__bf16*)(ws);
  __bf16* wT  = (__bf16*)(ws + (size_t)16 * 1024 * 1024);
  __bf16* qks = (__bf16*)(ws + (size_t)18 * 1024 * 1024);
  __bf16* vt  = (__bf16*)(ws + (size_t)34 * 1024 * 1024);
  float*  res = (float*) (ws + (size_t)50 * 1024 * 1024);

  deg_gate_kernel<<<B_ * N_, 256, 0, stream>>>(x, rel, wd, bd, xg);
  wq_transpose_kernel<<<(TDIM_ / 32) * (DIM_ / 32), 256, 0, stream>>>(wqkv, wT);
  qkv_kernel<<<B_ * N_ / 16, 256, 0, stream>>>(xg, wT, bqkv, qks, vt, res);
  attn_kernel<<<B_ * N_ / 16, 128, 0, stream>>>(qks, vt, res, rel, scale, out);
}
